// TreeCRF_20409684591087
// MI455X (gfx1250) — matrix operand order repaired and checked by reading, not compiled, not run on hardware
//
#include <hip/hip_runtime.h>
#include <hip/hip_bf16.h>
#include <math.h>

typedef __attribute__((ext_vector_type(16))) _Float16 v16h;
typedef __attribute__((ext_vector_type(8)))  float    v8f;

#define L 32

// ---------------------------------------------------------------------------
// Prep: rowmax[i] = max_j trans[i,j];  Bbuf holds exp(trans[i,j]-rowmax[i])
// pre-packed as WMMA B fragments in contiguous K order (2 tiles of 32 lanes x v16h):
//   tile t covers output states i = t*16 + (lane&15);
//   lane<16 holds K=j=0..15, lane>=16 holds K=j=16..31.
// ---------------------------------------------------------------------------
__global__ void crf_prep(const float* __restrict__ trans,
                         float* __restrict__ rowmax,
                         unsigned short* __restrict__ Bbuf) {
  const int tid = threadIdx.x;        // 0..63
  const int t   = tid >> 5;           // B tile (0 or 1)
  const int l   = tid & 31;           // lane slot within tile
  const int i   = t * 16 + (l & 15);  // output state (B column)

  float rm = -INFINITY;
  for (int j = 0; j < L; ++j) rm = fmaxf(rm, trans[i * L + j]);

  const int jbase = (l >> 4) * 16;
  v16h v;
#pragma unroll
  for (int k = 0; k < 16; ++k) {
    v[k] = (_Float16)__expf(trans[i * L + jbase + k] - rm);
  }
  ((v16h*)Bbuf)[t * 32 + l] = v;

  if (tid < L) {
    float m = -INFINITY;
    for (int j = 0; j < L; ++j) m = fmaxf(m, trans[tid * L + j]);
    rowmax[tid] = m;
  }
}

// Load one child's 32-float score row (this lane's 16-of-32 slice in contiguous
// K order k = 16h + s, as in Bbuf), compute the child max across the lane pair, and return
// exp(x - max) packed as the A-matrix v16h fragment.
__device__ __forceinline__ v16h load_child_frag(const float* __restrict__ s,
                                                int h, float& mC) {
  float xs[16];
#pragma unroll
  for (int q = 0; q < 2; ++q) {
    const float* p = s + h * 16 + q * 8;
    float4 u = *(const float4*)(p);
    float4 w = *(const float4*)(p + 4);
    xs[q * 8 + 0] = u.x; xs[q * 8 + 1] = u.y;
    xs[q * 8 + 2] = u.z; xs[q * 8 + 3] = u.w;
    xs[q * 8 + 4] = w.x; xs[q * 8 + 5] = w.y;
    xs[q * 8 + 6] = w.z; xs[q * 8 + 7] = w.w;
  }
  float m = xs[0];
#pragma unroll
  for (int k = 1; k < 16; ++k) m = fmaxf(m, xs[k]);
  mC = fmaxf(m, __shfl_xor(m, 16, 32));  // combine with partner lane (other 16 K's)
  v16h a;
#pragma unroll
  for (int k = 0; k < 16; ++k) a[k] = (_Float16)__expf(xs[k] - mC);
  return a;
}

// ---------------------------------------------------------------------------
// One tree level (level >= 16 parents). Each wave handles 16 parents:
//   A_left  = exp(left-child scores - mL)  (16x32 f16)
//   A_right = exp(right-child scores - mR) (16x32 f16)
// 4 x v_wmma_f32_16x16x32_f16 against the two B tiles give the child sums,
// then score = em + mL + mR + 2*rowmax + log(CL) + log(CR).
// ---------------------------------------------------------------------------
__global__ void crf_level_wmma(const float* __restrict__ childSrc,
                               const float* __restrict__ emissions,
                               float* __restrict__ scores,
                               const float* __restrict__ rowmax,
                               const unsigned short* __restrict__ Bbuf,
                               int level) {
  const int wave = threadIdx.x >> 5;
  const int lane = threadIdx.x & 31;
  const int pTileBase = (blockIdx.x * 8 + wave) * 16;   // 16 parents per wave
  if (pTileBase >= level) return;                        // wave-uniform exit

  const int M = lane & 15;   // A row / C column index
  const int h = lane >> 4;   // which K-half this lane carries

  const int parentNode = (level - 1) + pTileBase + M;
  const int leftNode   = 2 * parentNode + 1;
  const int rightNode  = leftNode + 1;

  // B tiles (states 0..15 and 16..31), pre-packed in B-matrix layout.
  const v16h b0 = ((const v16h*)Bbuf)[lane];
  const v16h b1 = ((const v16h*)Bbuf)[32 + lane];

  float mL, mR;
  const v16h aL = load_child_frag(childSrc + (size_t)leftNode * L,  h, mL);
  const v16h aR = load_child_frag(childSrc + (size_t)rightNode * L, h, mR);

  const v8f cz = {};
  v8f cL0 = __builtin_amdgcn_wmma_f32_16x16x32_f16(false, aL, false, b0, (short)0, cz, false, false);
  v8f cL1 = __builtin_amdgcn_wmma_f32_16x16x32_f16(false, aL, false, b1, (short)0, cz, false, false);
  v8f cR0 = __builtin_amdgcn_wmma_f32_16x16x32_f16(false, aR, false, b0, (short)0, cz, false, false);
  v8f cR1 = __builtin_amdgcn_wmma_f32_16x16x32_f16(false, aR, false, b1, (short)0, cz, false, false);

  const float rm0 = rowmax[M];
  const float rm1 = rowmax[M + 16];

#pragma unroll
  for (int r = 0; r < 8; ++r) {
    const int m = r + 8 * h;                 // C row held by this lane in vgpr r
    const float mLb = __shfl(mL, m, 32);     // per-child maxes live in lane m
    const float mRb = __shfl(mR, m, 32);
    const int pn = (level - 1) + pTileBase + m;
    const float base = mLb + mRb;
    const size_t off = (size_t)pn * L + M;
    scores[off]      = emissions[off]      + base + 2.0f * rm0 + __logf(cL0[r]) + __logf(cR0[r]);
    scores[off + 16] = emissions[off + 16] + base + 2.0f * rm1 + __logf(cL1[r]) + __logf(cR1[r]);
  }
}

// ---------------------------------------------------------------------------
// Tiny levels (8,4,2,1 parents): one thread per (parent, state).
// ---------------------------------------------------------------------------
__global__ void crf_level_small(const float* __restrict__ emissions,
                                const float* __restrict__ trans,
                                float* scores,
                                float* __restrict__ outRoot,
                                int level) {
  const int tid = threadIdx.x;
  if (tid >= level * L) return;
  const int n = tid >> 5;
  const int i = tid & 31;
  const int parentNode = (level - 1) + n;

  float acc = emissions[parentNode * L + i];
#pragma unroll
  for (int c = 0; c < 2; ++c) {
    const float* s = scores + (size_t)(2 * parentNode + 1 + c) * L;
    float m = -INFINITY;
    for (int j = 0; j < L; ++j) m = fmaxf(m, trans[i * L + j] + s[j]);
    float sum = 0.0f;
    for (int j = 0; j < L; ++j) sum += __expf(trans[i * L + j] + s[j] - m);
    acc += m + __logf(sum);
  }
  if (level == 1) outRoot[i] = acc;
  else            scores[parentNode * L + i] = acc;
}

extern "C" void kernel_launch(void* const* d_in, const int* in_sizes, int n_in,
                              void* d_out, int out_size, void* d_ws, size_t ws_size,
                              hipStream_t stream) {
  const float* emissions = (const float*)d_in[0];
  const float* trans     = (const float*)d_in[1];
  float* out = (float*)d_out;

  float* ws = (float*)d_ws;
  float*          rowmax = ws;                          // 32 floats
  unsigned short* Bbuf   = (unsigned short*)(ws + 32);  // 1024 f16 (2 KB), 128B aligned
  float*          scores = ws + 1024;                   // (n_leaves-1) x 32 floats (~16.8 MB)

  const int n_nodes  = in_sizes[0] / L;                 // 262143
  const int n_leaves = (n_nodes + 1) / 2;               // 131072

  crf_prep<<<1, 64, 0, stream>>>(trans, rowmax, Bbuf);

  // WMMA levels: 65536 down to 16 parents. First level reads leaf scores
  // straight out of `emissions` (leaf scores == emission rows, heap-indexed).
  for (int level = n_leaves / 2; level >= 16; level >>= 1) {
    const float* childSrc = (level == n_leaves / 2) ? emissions : scores;
    const int blocks = (level + 127) / 128;             // 8 waves x 16 parents per block
    crf_level_wmma<<<blocks, 256, 0, stream>>>(childSrc, emissions, scores,
                                               rowmax, Bbuf, level);
  }

  // Scalar tail: levels 8,4,2,1 (root written to d_out).
  for (int level = 8; level >= 1; level >>= 1) {
    crf_level_small<<<1, 256, 0, stream>>>(emissions, trans, scores, out, level);
  }
}
